// BondOutputModule_49572512530564
// MI455X (gfx1250) — compile-verified, matrix operand order checked by reading, not run on hardware
//
#include <hip/hip_runtime.h>
#include <hip/hip_bf16.h>

#define N_BOND     100000
#define DIM        192
#define N_ETYPES   36
#define E_PER_TYPE 30000
#define BATCH      256

typedef __attribute__((ext_vector_type(16))) _Float16 v16h;
typedef __attribute__((ext_vector_type(8)))  _Float16 v8h;
typedef __attribute__((ext_vector_type(8)))  float    v8f;

// -------------------------------------------------------------------------
// Kernel 1: p[n] = dot(h[n, 0:192], w[0:192]) via V_WMMA_F32_16X16X32_F16.
// Each wave computes 16 nodes; block = 4 waves = 64 nodes. h rows are staged
// into LDS as f16 with coalesced float4 global loads, then each lane builds
// its A operand with two 16-byte LDS reads per 32-wide K block. B holds w
// replicated into all 16 columns, so every accumulator column equals p.
// -------------------------------------------------------------------------
#define NODES_PER_BLOCK 64
#define ROW_PAD 8   // 200 halves = 400B row stride (16B aligned, conflict-light)

__global__ __launch_bounds__(128) void bond_project_wmma(
    const float* __restrict__ h, const float* __restrict__ w,
    float* __restrict__ p, int n_nodes)
{
    __shared__ __align__(16) _Float16 sh[NODES_PER_BLOCK][DIM + ROW_PAD];

    const int tid        = threadIdx.x;
    const int block_base = blockIdx.x * NODES_PER_BLOCK;

    // Stage 64 rows of h into LDS as f16 (coalesced float4 loads, clamp tail).
    for (int i = tid; i < NODES_PER_BLOCK * (DIM / 4); i += 128) {
        int row  = i / (DIM / 4);
        int c4   = i % (DIM / 4);
        int node = block_base + row;
        if (node >= n_nodes) node = n_nodes - 1;
        const float4 v = ((const float4*)(h + (size_t)node * DIM))[c4];
        sh[row][c4 * 4 + 0] = (_Float16)v.x;
        sh[row][c4 * 4 + 1] = (_Float16)v.y;
        sh[row][c4 * 4 + 2] = (_Float16)v.z;
        sh[row][c4 * 4 + 3] = (_Float16)v.w;
    }

    const int lane = tid & 31;
    const int wave = tid >> 5;
    // 16-bit A/B element->K mapping (ISA 7.12.2): lanes 0-15 hold K {0..7,16..23},
    // lanes 16-31 hold K {8..15,24..31} within each 32-wide K block.
    const int lo = (lane < 16) ? 0 : 8;

    // B operand: w replicated into every column (K x 16), per 32-K block.
    v16h bmat[6];
    #pragma unroll
    for (int kb = 0; kb < 6; ++kb) {
        const int kbase = kb * 32;
        #pragma unroll
        for (int e = 0; e < 8; ++e) {
            bmat[kb][e]     = (_Float16)w[kbase + lo + e];
            bmat[kb][e + 8] = (_Float16)w[kbase + 16 + lo + e];
        }
    }
    __syncthreads();

    const int r = wave * 16 + (lane & 15);   // LDS row providing this lane's M row
    v8f c = {};
    #pragma unroll
    for (int kb = 0; kb < 6; ++kb) {
        const int kbase = kb * 32;
        v8h a_lo = *(const v8h*)&sh[r][kbase + lo];        // K = kbase+lo .. +7
        v8h a_hi = *(const v8h*)&sh[r][kbase + 16 + lo];   // K = kbase+16+lo .. +7
        v16h a = __builtin_shufflevector(a_lo, a_hi,
                                         0,1,2,3,4,5,6,7,8,9,10,11,12,13,14,15);
        c = __builtin_amdgcn_wmma_f32_16x16x32_f16(
                /*neg_a=*/false, a, /*neg_b=*/false, bmat[kb],
                /*c_mod=*/(short)0, c, /*reuse_a=*/false, /*reuse_b=*/false);
    }

    // C/D layout: VGPR i, lane 0  -> (M=i,   N=0); lane 16 -> (M=8+i, N=0).
    const int wave_base = block_base + wave * 16;
    if (wave_base < n_nodes) {
        float cf[8];
        #pragma unroll
        for (int i = 0; i < 8; ++i) cf[i] = c[i];
        if (lane == 0) {
            #pragma unroll
            for (int i = 0; i < 8; ++i) p[wave_base + i] = cf[i];
        } else if (lane == 16) {
            #pragma unroll
            for (int i = 0; i < 8; ++i) p[wave_base + 8 + i] = cf[i];
        }
    }
}

// -------------------------------------------------------------------------
// Kernel 2: per edge type, segment-sum p[src[e]] by graph id into sums[t,b].
// edge_seg is sorted within a type, so each thread run-length accumulates a
// contiguous chunk and fires one LDS float atomic per segment boundary.
// -------------------------------------------------------------------------
__global__ __launch_bounds__(512) void edge_segsum(
    const float* __restrict__ p, const int* __restrict__ edge_src,
    const int* __restrict__ edge_seg, float* __restrict__ sums)
{
    const int t = blockIdx.x;                      // edge type
    __shared__ float acc[BATCH];
    for (int i = threadIdx.x; i < BATCH; i += 512) acc[i] = 0.0f;
    __syncthreads();

    const int* src = edge_src + t * E_PER_TYPE;
    const int* seg = edge_seg + t * E_PER_TYPE;
    const int per = (E_PER_TYPE + 511) / 512;      // 59 edges per thread
    const int e0  = threadIdx.x * per;
    const int e1  = (e0 + per < E_PER_TYPE) ? (e0 + per) : E_PER_TYPE;

    float run = 0.0f;
    int   cur = -1;
    for (int e = e0; e < e1; ++e) {
        const int   s = seg[e];
        const float v = p[src[e]];                 // 400KB table, L2 resident
        if (s != cur) {
            if (cur >= 0) atomicAdd(&acc[cur], run);
            cur = s; run = v;
        } else {
            run += v;
        }
    }
    if (cur >= 0) atomicAdd(&acc[cur], run);
    __syncthreads();

    for (int i = threadIdx.x; i < BATCH; i += 512)
        sums[t * BATCH + i] = acc[i];
}

// -------------------------------------------------------------------------
// Kernel 3: per graph b, masked softmax over the 36 edge types.
// -------------------------------------------------------------------------
__global__ __launch_bounds__(128) void masked_softmax(
    const float* __restrict__ sums, const unsigned char* __restrict__ mask,
    float* __restrict__ out)
{
    const int b = blockIdx.x * blockDim.x + threadIdx.x;
    if (b >= BATCH) return;

    float v[N_ETYPES];
    float mx = -3.402823466e+38f;
    #pragma unroll
    for (int t = 0; t < N_ETYPES; ++t) {
        const float s   = sums[t * BATCH + b];     // coalesced across b
        const float val = mask[b * N_ETYPES + t] ? -1.0e9f : s;
        v[t] = val;
        mx   = fmaxf(mx, val);
    }
    float sum = 0.0f;
    #pragma unroll
    for (int t = 0; t < N_ETYPES; ++t) {
        const float e = __expf(v[t] - mx);
        v[t] = e;
        sum += e;
    }
    const float inv = 1.0f / sum;
    #pragma unroll
    for (int t = 0; t < N_ETYPES; ++t)
        out[b * N_ETYPES + t] = v[t] * inv;
}

// -------------------------------------------------------------------------
extern "C" void kernel_launch(void* const* d_in, const int* in_sizes, int n_in,
                              void* d_out, int out_size, void* d_ws, size_t ws_size,
                              hipStream_t stream)
{
    const float*         h    = (const float*)d_in[0];          // [100000,192]
    const float*         w    = (const float*)d_in[1];          // [192,1]
    const int*           esrc = (const int*)d_in[2];            // [36,30000]
    const int*           eseg = (const int*)d_in[3];            // [36,30000]
    const unsigned char* mask = (const unsigned char*)d_in[4];  // [256,36] bool
    float*               out  = (float*)d_out;                  // [256,36]

    float* p    = (float*)d_ws;                    // N_BOND floats
    float* sums = p + ((N_BOND + 63) & ~63);       // N_ETYPES*BATCH floats

    const int nblocks = (N_BOND + NODES_PER_BLOCK - 1) / NODES_PER_BLOCK;
    bond_project_wmma<<<nblocks, 128, 0, stream>>>(h, w, p, N_BOND);
    edge_segsum<<<N_ETYPES, 512, 0, stream>>>(p, esrc, eseg, sums);
    masked_softmax<<<(BATCH + 127) / 128, 128, 0, stream>>>(sums, mask, out);
}
